// WindowAttention_10075993277131
// MI455X (gfx1250) — matrix operand order repaired and checked by reading, not compiled, not run on hardware
//
#include <hip/hip_runtime.h>
#include <hip/hip_bf16.h>
#include <float.h>

// ---- CDNA5 wave32 WMMA types ----
typedef __attribute__((ext_vector_type(16))) __bf16   v16bf;
typedef __attribute__((ext_vector_type(8)))  float    v8f;
typedef __attribute__((ext_vector_type(8)))  unsigned v8u;

#define WWH   7
#define WWW   7
#define NTOK  49
#define CCH   256
#define NH    8
#define HDM   32
#define BWIN  4096
#define NWIN  64
#define ROWS  (BWIN * NTOK)   // 200704 (divisible by 64)

static __device__ __forceinline__ unsigned pack_bf16(float a, float b) {
    __bf16 ha = (__bf16)a, hb = (__bf16)b;
    unsigned short ua = __builtin_bit_cast(unsigned short, ha);
    unsigned short ub = __builtin_bit_cast(unsigned short, hb);
    return ((unsigned)ub << 16) | (unsigned)ua;
}

static __device__ __forceinline__ v16bf frag_cast(v8u u) {
    return __builtin_bit_cast(v16bf, u);
}

static __device__ __forceinline__ v8f wmma_bf16(v16bf a, v16bf b, v8f c) {
    // D(f32 16x16) = A(bf16 16x32) * B(bf16 32x16) + C
    return __builtin_amdgcn_wmma_f32_16x16x32_bf16(false, a, false, b, (short)0, c,
                                                   false, false);
}

// ---------------------------------------------------------------------------
// Tiled GEMM: Out[M x ncols] = A[M x 256] * B[256 x ncols] (+ bias)
// A: fp32 or bf16 (converted/packed to bf16 in LDS), B: fp32 weights -> bf16.
// Block: 256 threads = 8 waves as 4(M) x 2(N); block tile 64 x 128.
// Each wave: one 16-row M tile x four 16-col N tiles, K-steps of 32.
// Next K-step staged with global_prefetch to overlap HBM/L2 latency.
// ---------------------------------------------------------------------------
template<bool A_BF16, bool ADD_BIAS, bool OUT_F32>
__global__ __launch_bounds__(256)
void gemm_wmma(const void* __restrict__ Ap, const float* __restrict__ B,
               const float* __restrict__ bias, void* __restrict__ Out,
               int lda, int ldb, int ncols)
{
    __shared__ unsigned lA[64][16];  // [m][k/2] packed bf16 pairs (4 KB)
    __shared__ unsigned lB[32][64];  // [k][n/2] packed bf16 pairs (8 KB)

    const int tid   = threadIdx.x;
    const int lane  = tid & 31;
    const int wave  = tid >> 5;
    const int waveM = wave >> 1;          // 0..3
    const int waveN = wave & 1;           // 0..1
    const int half  = lane >> 4;
    const int l16   = lane & 15;
    const long mBase = (long)blockIdx.x * 64;
    const long nBase = (long)blockIdx.y * 128;

    v8f acc[4] = {};

    for (int k0 = 0; k0 < 256; k0 += 32) {
        // Stage A tile 64x32 (1024 packed pairs, 4 per thread)
        #pragma unroll
        for (int i = 0; i < 4; i++) {
            int s = tid + i * 256;
            int m = s >> 4, kp = s & 15;
            long row = mBase + m;
            long col = k0 + kp * 2;
            unsigned v;
            if (A_BF16) {
                v = ((const unsigned*)Ap)[(row * (long)lda + col) >> 1];
            } else {
                const float* Af = (const float*)Ap;
                v = pack_bf16(Af[row * (long)lda + col],
                              Af[row * (long)lda + col + 1]);
            }
            lA[m][kp] = v;
            // Prefetch next K-step of A (gfx1250 global_prefetch_b8)
            if (k0 + 32 < 256) {
                if (A_BF16)
                    __builtin_prefetch(((const unsigned*)Ap) +
                                       ((row * (long)lda + col + 32) >> 1), 0, 1);
                else
                    __builtin_prefetch(((const float*)Ap) +
                                       (row * (long)lda + col + 32), 0, 1);
            }
        }
        // Stage B tile 32x128 (2048 packed pairs, 8 per thread)
        #pragma unroll
        for (int i = 0; i < 8; i++) {
            int s = tid + i * 256;
            int k = s >> 6, np = s & 63;
            long row = k0 + k;
            long col = nBase + np * 2;
            lB[k][np] = pack_bf16(B[row * (long)ldb + col],
                                  B[row * (long)ldb + col + 1]);
            if (k0 + 32 < 256)
                __builtin_prefetch(B + ((row + 32) * (long)ldb + col), 0, 1);
        }
        __syncthreads();

        // A fragment: lane -> row m, VGPR v -> K pair
        v8u au;
        #pragma unroll
        for (int v = 0; v < 8; v++) {
            int k = ((v & 4) << 2) | (half << 3) | ((v & 3) << 1);
            au[v] = lA[waveM * 16 + l16][k >> 1];
        }
        v16bf afrag = frag_cast(au);

        #pragma unroll
        for (int nt = 0; nt < 4; nt++) {
            // B fragment: lane -> n col, VGPR v -> K pair
            int n0 = waveN * 64 + nt * 16;
            v8u bu;
            #pragma unroll
            for (int v = 0; v < 8; v++) {
                int k = ((v & 4) << 2) | (half << 3) | ((v & 3) << 1);
                int sh = (l16 & 1) * 16;
                unsigned w0 = lB[k][(n0 + l16) >> 1], w1 = lB[k + 1][(n0 + l16) >> 1];
                bu[v] = ((w0 >> sh) & 0xffffu) | (((w1 >> sh) & 0xffffu) << 16);
            }
            acc[nt] = wmma_bf16(afrag, frag_cast(bu), acc[nt]);
        }
        __syncthreads();
    }

    // Epilogue: D layout lane -> n = lane%16, VGPR r -> m = r + 8*half
    #pragma unroll
    for (int nt = 0; nt < 4; nt++) {
        #pragma unroll
        for (int r = 0; r < 8; r++) {
            long m = mBase + waveM * 16 + half * 8 + r;
            long n = nBase + waveN * 64 + nt * 16 + l16;
            float val = acc[nt][r];
            if (ADD_BIAS) val += bias[n];
            if (OUT_F32) ((float*)Out)[m * (long)ncols + n] = val;
            else         ((__bf16*)Out)[m * (long)ncols + n] = (__bf16)val;
        }
    }
}

// ---------------------------------------------------------------------------
// Attention: one block (128 thr = 4 waves) per (window b, head h).
// Stage q/v token-major and K pre-transposed+packed (WMMA-B layout) in LDS,
// then: S[64x64 padded] = q k^T via WMMA (q fragment direct u32 LDS reads, k fragment merged from 16-bit halves),
// fp32 softmax (LSA diag mask, inline rel-pos bias, shift mask),
// O[64x32] = P V via WMMA.
// qkv buffer bf16 [ROWS][768], cols: q = h*32, k = 256+h*32, v = 512+h*32.
// ---------------------------------------------------------------------------
__global__ __launch_bounds__(128)
void attn_wmma(const __bf16* __restrict__ qkv, const float* __restrict__ mask,
               const float* __restrict__ rpb, const float* __restrict__ tptr,
               __bf16* __restrict__ outbf)
{
    __shared__ float    S[64][64];    // fp32 scores (16 KB)
    __shared__ unsigned P[64][32];    // packed bf16 probs (8 KB)
    __shared__ unsigned Qb[64][16];   // q token-major, packed hd pairs (4 KB)
    __shared__ unsigned KT[32][32];   // k^T: [hd][token-pair] packed (4 KB)
    __shared__ unsigned Vb[64][16];   // v token-major, packed hd pairs (4 KB)

    const int tid  = threadIdx.x;
    const int lane = tid & 31;
    const int wave = tid >> 5;               // 0..3 -> M tile
    const int b    = blockIdx.x >> 3;
    const int h    = blockIdx.x & 7;
    const int half = lane >> 4;
    const int l16  = lane & 15;
    const int m0   = wave * 16;

    const unsigned* g32 = (const unsigned*)qkv;
    const long rowu = ((long)b * NTOK) * 384;   // u32 index of window base row

    // ---- Stage q and v (2 x 64 x 16 u32; rows >= 49 clamp to 48) ----
    #pragma unroll
    for (int i = 0; i < 16; i++) {
        int s   = tid + i * 128;        // 0..2047
        int sel = s >> 10;              // 0:q 1:v
        int tok = (s >> 4) & 63;
        int p   = s & 15;
        int tokc = (tok > NTOK - 1) ? (NTOK - 1) : tok;
        unsigned v = g32[rowu + (long)tokc * 384 + sel * 128 + h * 16 + p];
        if (sel == 0) Qb[tok][p] = v;
        else          Vb[tok][p] = v;
    }
    // ---- Stage K transposed+packed: KT[hd][tp] = (K[2tp+1][hd], K[2tp][hd]) ----
    #pragma unroll
    for (int i = 0; i < 4; i++) {
        int s  = tid + i * 128;         // 0..511
        int tp = s >> 4;                // token pair 0..31
        int p  = s & 15;                // hd pair 0..15
        int t0 = 2 * tp;     if (t0 > NTOK - 1) t0 = NTOK - 1;
        int t1 = 2 * tp + 1; if (t1 > NTOK - 1) t1 = NTOK - 1;
        unsigned a  = g32[rowu + (long)t0 * 384 + 128 + h * 16 + p];
        unsigned b2 = g32[rowu + (long)t1 * 384 + 128 + h * 16 + p];
        KT[2 * p][tp]     = (b2 << 16) | (a & 0xffffu);
        KT[2 * p + 1][tp] = (b2 & 0xffff0000u) | (a >> 16);
    }
    __syncthreads();

    // ---- S = q k^T (per wave: 16 rows x 64 cols, K = HD = 32) ----
    {
        v8u au;
        #pragma unroll
        for (int v = 0; v < 8; v++) {
            int k = ((v & 4) << 2) | (half << 3) | ((v & 3) << 1);
            au[v] = Qb[m0 + l16][k >> 1];
        }
        v16bf qf = frag_cast(au);

        #pragma unroll
        for (int nt = 0; nt < 4; nt++) {
            v8u bu;
            #pragma unroll
            for (int v = 0; v < 8; v++) {
                int k = ((v & 4) << 2) | (half << 3) | ((v & 3) << 1);
                int sh = (l16 & 1) * 16;
                unsigned w0 = KT[k][(nt * 16 + l16) >> 1], w1 = KT[k + 1][(nt * 16 + l16) >> 1];
                bu[v] = ((w0 >> sh) & 0xffffu) | (((w1 >> sh) & 0xffffu) << 16);
            }
            v8f sacc = {};
            sacc = wmma_bf16(qf, frag_cast(bu), sacc);
            #pragma unroll
            for (int r = 0; r < 8; r++)
                S[m0 + half * 8 + r][nt * 16 + l16] = sacc[r];
        }
    }
    __syncthreads();

    // ---- Row softmax (threads 0..63, one row each) ----
    if (tid < 64) {
        const int m = tid;
        if (m >= NTOK) {
            #pragma unroll
            for (int np = 0; np < 32; np++) P[m][np] = 0u;
        } else {
            const float scale = __expf(tptr[0]);
            const float* mrow = mask + ((long)(b & (NWIN - 1))) * NTOK * NTOK + m * NTOK;
            const int mi = m / WWW, mj = m % WWW;
            float vals[NTOK];
            float maxv = -FLT_MAX;
            #pragma unroll
            for (int n = 0; n < NTOK; n++) {
                float v;
                if (n == m) {
                    v = -FLT_MAX;  // LSA: mask diagonal with most-negative float
                } else {
                    int ni = n / WWW, nj = n % WWW;
                    int idx = (mi - ni + WWW - 1) * (2 * WWW - 1) + (mj - nj + WWW - 1);
                    v = S[m][n] * scale + rpb[idx * NH + h] + mrow[n];
                }
                vals[n] = v;
                maxv = fmaxf(maxv, v);
            }
            float sum = 0.f;
            #pragma unroll
            for (int n = 0; n < NTOK; n++) {
                float e = __expf(vals[n] - maxv);
                vals[n] = e;
                sum += e;
            }
            float inv = 1.0f / sum;
            #pragma unroll
            for (int np = 0; np < 32; np++) {
                int n0 = 2 * np, n1 = 2 * np + 1;
                float p0 = (n0 < NTOK) ? vals[n0] * inv : 0.f;
                float p1 = (n1 < NTOK) ? vals[n1] * inv : 0.f;
                P[m][np] = pack_bf16(p0, p1);
            }
        }
    }
    __syncthreads();

    // ---- O = P V (per wave: 16 rows x 32 cols, K = 64 in 2 steps) ----
    {
        v8f o0 = {}, o1 = {};
        #pragma unroll
        for (int s = 0; s < 2; s++) {
            v8u au;
            #pragma unroll
            for (int v = 0; v < 8; v++) {
                int k = (((v & 4) << 2) | (half << 3) | ((v & 3) << 1)) + 32 * s;
                au[v] = P[m0 + l16][k >> 1];
            }
            v16bf pf = frag_cast(au);
            int tok = 32 * s + lane;          // P cols >= 49 are zero, V rows clamped
            v8u bu0, bu1;
            #pragma unroll
            for (int v = 0; v < 8; v++) {
                int k = (((v & 4) << 2) | (half << 3) | ((v & 3) << 1)) + 32 * s;
                int sh = (l16 & 1) * 16;
                unsigned a0 = Vb[k][l16 >> 1], a1 = Vb[k + 1][l16 >> 1];
                unsigned c0 = Vb[k][8 + (l16 >> 1)], c1 = Vb[k + 1][8 + (l16 >> 1)];
                bu0[v] = ((a0 >> sh) & 0xffffu) | (((a1 >> sh) & 0xffffu) << 16);
                bu1[v] = ((c0 >> sh) & 0xffffu) | (((c1 >> sh) & 0xffffu) << 16);
            }
            o0 = wmma_bf16(pf, frag_cast(bu0), o0);
            o1 = wmma_bf16(pf, frag_cast(bu1), o1);
        }
        #pragma unroll
        for (int r = 0; r < 8; r++) {
            int m = m0 + half * 8 + r;
            if (m < NTOK) {
                long rowb = ((long)(b * NTOK + m)) * CCH + h * HDM;
                outbf[rowb + l16]      = (__bf16)o0[r];
                outbf[rowb + 16 + l16] = (__bf16)o1[r];
            }
        }
    }
}

extern "C" void kernel_launch(void* const* d_in, const int* in_sizes, int n_in,
                              void* d_out, int out_size, void* d_ws, size_t ws_size,
                              hipStream_t stream)
{
    const float* x      = (const float*)d_in[0];
    const float* mask   = (const float*)d_in[1];
    const float* qkv_w  = (const float*)d_in[2];
    const float* proj_w = (const float*)d_in[3];
    const float* proj_b = (const float*)d_in[4];
    const float* rpb    = (const float*)d_in[5];
    const float* temp   = (const float*)d_in[6];
    (void)in_sizes; (void)n_in; (void)out_size; (void)ws_size;

    // Workspace: qkv (bf16, ROWS x 768) then attention output (bf16, ROWS x 256)
    __bf16* qkv_bf = (__bf16*)d_ws;
    __bf16* ao_bf  = qkv_bf + (size_t)ROWS * 768;

    // 1) QKV projection GEMM (fp32 in -> bf16 WMMA -> bf16 out)
    gemm_wmma<false, false, false><<<dim3(ROWS / 64, 768 / 128), dim3(256), 0, stream>>>(
        x, qkv_w, nullptr, qkv_bf, CCH, 768, 768);

    // 2) Window attention per (window, head)
    attn_wmma<<<dim3(BWIN * NH), dim3(128), 0, stream>>>(qkv_bf, mask, rpb, temp, ao_bf);

    // 3) Output projection GEMM (+ bias, fp32 out)
    gemm_wmma<true, true, true><<<dim3(ROWS / 64, 256 / 128), dim3(256), 0, stream>>>(
        ao_bf, proj_w, proj_b, (float*)d_out, CCH, CCH, CCH);
}
